// CSNN_26199300506035
// MI455X (gfx1250) — matrix operand order repaired and checked by reading, not compiled, not run on hardware
//
#include <hip/hip_runtime.h>
#include <hip/hip_bf16.h>

typedef __attribute__((ext_vector_type(16))) _Float16 v16h;
typedef __attribute__((ext_vector_type(8)))  float    v8f;
typedef __attribute__((ext_vector_type(2)))  float    v2f;

#define BETA 0.5f

// Problem dims (fixed by reference)
constexpr int Bn  = 64;    // batch
constexpr int Tn  = 32;    // time steps
constexpr int L0  = 2048;  // input length
constexpr int L1  = 1023;  // after conv1+pool
constexpr int L2  = 511;   // after conv2+pool
constexpr int SROW  = 32;  // staging channels (f16) per position
constexpr int SROWS = 1048; // 1 front halo + 1023 + tail halo (rounded)

// ---- output layout (floats), tuple order: spk0,spk1,spk2,spk3,mem0,mem1,mem2,mem3 ----
constexpr size_t S0 = (size_t)32 * L1;        // 32736
constexpr size_t S1 = (size_t)64 * L2;        // 32704
constexpr size_t TB = (size_t)Tn * Bn;        // 2048
constexpr size_t BASE_SPK0 = 0;
constexpr size_t BASE_SPK1 = BASE_SPK0 + TB * S0;
constexpr size_t BASE_SPK2 = BASE_SPK1 + TB * S1;
constexpr size_t BASE_SPK3 = BASE_SPK2 + TB * 256;
constexpr size_t BASE_MEM0 = BASE_SPK3 + TB * 2;
constexpr size_t BASE_MEM1 = BASE_MEM0 + TB * S0;
constexpr size_t BASE_MEM2 = BASE_MEM1 + TB * S1;
constexpr size_t BASE_MEM3 = BASE_MEM2 + TB * 256;

// ---- workspace layout (bytes) ----
constexpr size_t SZ_MEM0  = (size_t)Bn * 32 * L1;      // floats
constexpr size_t SZ_MEM1  = (size_t)Bn * 64 * L2;      // floats
constexpr size_t SZ_MEM2  = (size_t)Bn * 256;          // floats
constexpr size_t SZ_MEM3  = (size_t)Bn * 2;            // floats
constexpr size_t SZ_SPK0H = (size_t)Bn * SROWS * SROW; // halves
constexpr size_t alignup(size_t x) { return (x + 255) & ~(size_t)255; }
constexpr size_t OFF_MEM0  = 0;
constexpr size_t OFF_MEM1  = alignup(OFF_MEM0 + SZ_MEM0 * 4);
constexpr size_t OFF_MEM2  = alignup(OFF_MEM1 + SZ_MEM1 * 4);
constexpr size_t OFF_MEM3  = alignup(OFF_MEM2 + SZ_MEM2 * 4);
constexpr size_t OFF_SPK0H = alignup(OFF_MEM3 + SZ_MEM3 * 4);
constexpr size_t ZERO_END  = OFF_SPK0H + SZ_SPK0H * 2;   // persistent state, zero each launch
constexpr size_t OFF_W2P   = alignup(ZERO_END);          // packed f16 conv2 weights (6144 halves)
constexpr size_t OFF_FEAT  = alignup(OFF_W2P + 6144 * 2); // feat spike-count accumulators [Bn*128]

__global__ void zero_kernel(float* p, size_t n) {
    size_t i = (size_t)blockIdx.x * blockDim.x + threadIdx.x;
    if (i < n) p[i] = 0.f;
}

// Pack conv2_w [64,32,3] f32 -> f16 A fragments; global K = tap*32 + ic.
// A 16x32 in contiguous k order (matches B fragment, not AMD's table order): lane l (M=l&15), halves j -> localK=16*(l>>4)+j.
__global__ void pack_w2_kernel(const float* __restrict__ w2, _Float16* __restrict__ w2p) {
    int idx = blockIdx.x * blockDim.x + threadIdx.x;
    if (idx >= 4 * 3 * 32 * 16) return;
    int j    = idx & 15;
    int lane = (idx >> 4) & 31;
    int kc   = (idx >> 9) % 3;
    int mt   = idx / (16 * 32 * 3);
    int lg   = lane >> 4;
    int localK = 16 * lg + j;
    int ic = localK;                 // chunk kc == tap
    int oc = 16 * mt + (lane & 15);
    w2p[idx] = (_Float16)w2[(oc * 32 + ic) * 3 + kc];
}

// ---- Layer 1: conv1(4->32,k3,pad1) + maxpool(3,2) + LIF; writes recs + f16 staging.
//      Block 0 additionally zeroes the feat accumulators for this step. ----
__global__ void layer1_kernel(const float* __restrict__ x, const float* __restrict__ w1,
                              const float* __restrict__ b1, float* __restrict__ mem0,
                              _Float16* __restrict__ spk0h, float* __restrict__ feat,
                              float* __restrict__ out, const float* __restrict__ thr, int t) {
    if (blockIdx.x == 0) {
        for (int i = threadIdx.x; i < Bn * 128; i += 256) feat[i] = 0.f;
    }
    int idx = blockIdx.x * blockDim.x + threadIdx.x;
    if (idx >= Bn * 32 * L1) return;
    int j  = idx % L1;
    int oc = (idx / L1) % 32;
    int b  = idx / (L1 * 32);
    const float* xb = x + ((size_t)(b * Tn + t) * 4) * L0;
    float wv[12];
#pragma unroll
    for (int c = 0; c < 4; c++)
#pragma unroll
        for (int k = 0; k < 3; k++) wv[c * 3 + k] = w1[(oc * 4 + c) * 3 + k];
    float bias = b1[oc];
    float mx = -INFINITY;
#pragma unroll
    for (int u = 0; u < 3; u++) {
        int p = 2 * j + u;
        float s = bias;
#pragma unroll
        for (int c = 0; c < 4; c++)
#pragma unroll
            for (int k = 0; k < 3; k++) {
                int q = p + k - 1;
                float xv = (q >= 0 && q < L0) ? xb[c * L0 + q] : 0.f;
                s = fmaf(wv[c * 3 + k], xv, s);
            }
        mx = fmaxf(mx, s);
    }
    float th = thr[0];
    float m = mem0[idx];
    float reset = (m - th > 0.f) ? 1.f : 0.f;
    m = BETA * m + mx - reset * th;
    float spk = (m - th > 0.f) ? 1.f : 0.f;
    mem0[idx] = m;
    size_t rec = ((size_t)t * Bn + b) * S0 + (size_t)oc * L1 + j;
    out[BASE_SPK0 + rec] = spk;
    out[BASE_MEM0 + rec] = m;
    spk0h[((size_t)b * SROWS + (j + 1)) * SROW + oc] = (_Float16)spk;
}

// ---- Layer 2: conv2(32->64,k3,pad1) as implicit-GEMM WMMA + maxpool(3,2) + LIF.
//      Also accumulates adaptive-avg-pool spike counts into feat (exact: 0/1 sums). ----
// grid = Bn*8 blocks; block = 8 waves; pooled chunk of 64 per block.
__global__ void __launch_bounds__(256) layer2_kernel(
        const _Float16* __restrict__ spk0h, const _Float16* __restrict__ w2p,
        const float* __restrict__ b2, float* __restrict__ mem1, float* __restrict__ feat,
        float* __restrict__ out, const float* __restrict__ thr, int t) {
    __shared__ float convs[64 * 144];
    int b   = blockIdx.x >> 3;
    int blk = blockIdx.x & 7;
    int jb  = blk * 64;
    int pbase0 = 2 * jb;
    int lane = threadIdx.x & 31;
    int wv   = threadIdx.x >> 5;
    int lg   = lane >> 4;
    int lm   = lane & 15;
    const _Float16* srcB = spk0h + (size_t)b * SROWS * SROW;
    for (int tile = wv; tile < 9; tile += 8) {       // wave0 also covers overhang tile
        int pbase = pbase0 + 16 * tile;
        v8f acc[4] = {};
#pragma unroll
        for (int kc = 0; kc < 3; kc++) {
            // B fragment: lane -> position pbase+lm, 16 contiguous channels at 16*lg
            v16h bfrag = *(const v16h*)(srcB + (size_t)(pbase + lm + kc) * SROW + 16 * lg);
#pragma unroll
            for (int m = 0; m < 4; m++) {
                v16h afrag = *(const v16h*)(w2p + ((m * 3 + kc) * 32 + lane) * 16);
                acc[m] = __builtin_amdgcn_wmma_f32_16x16x32_f16(
                    false, afrag, false, bfrag, (short)0, acc[m], false, false);
            }
        }
#pragma unroll
        for (int m = 0; m < 4; m++)
#pragma unroll
            for (int r = 0; r < 8; r++) {
                int oc = 16 * m + r + 8 * lg;
                convs[oc * 144 + 16 * tile + lm] = acc[m][r] + b2[oc];
            }
    }
    __syncthreads();
    float th = thr[1];
    for (int idx = threadIdx.x; idx < 64 * 64; idx += 256) {
        int oc = idx >> 6;
        int jj = idx & 63;
        int j = jb + jj;
        if (j >= L2) continue;
        float c0 = convs[oc * 144 + 2 * jj];
        float c1 = convs[oc * 144 + 2 * jj + 1];
        float c2 = convs[oc * 144 + 2 * jj + 2];
        float cur = fmaxf(c0, fmaxf(c1, c2));
        size_t mi = ((size_t)b * 64 + oc) * L2 + j;
        float m = mem1[mi];
        float reset = (m - th > 0.f) ? 1.f : 0.f;
        m = BETA * m + cur - reset * th;
        float spk = (m - th > 0.f) ? 1.f : 0.f;
        mem1[mi] = m;
        size_t rec = ((size_t)t * Bn + b) * S1 + (size_t)oc * L2 + j;
        out[BASE_SPK1 + rec] = spk;
        out[BASE_MEM1 + rec] = m;
        // adaptive_avg_pool1d(2) windows: [0,256) and [255,511); sums of 0/1 are exact
        if (spk > 0.f) {
            if (j < 256)  atomicAdd(&feat[b * 128 + oc * 2], 1.0f);
            if (j >= 255) atomicAdd(&feat[b * 128 + oc * 2 + 1], 1.0f);
        }
    }
}

// ---- Fused head: FC1 (64x256x128, f32 WMMA 16x16x4) + LIF + FC2 (64x2x256) + LIF ----
// Single block, 8 waves; spk2 staged through LDS as f16 (0/1 exact).
__global__ void __launch_bounds__(256) fc_fused_kernel(
        const float* __restrict__ feat, const float* __restrict__ fw1,
        const float* __restrict__ fb1, const float* __restrict__ fw2,
        const float* __restrict__ fb2, float* __restrict__ mem2, float* __restrict__ mem3,
        float* __restrict__ out, const float* __restrict__ thr, int t) {
    __shared__ _Float16 spk2s[64 * 256];
    int lane = threadIdx.x & 31, wv = threadIdx.x >> 5;
    int lg = lane >> 4, lm = lane & 15;
    float th2 = thr[2];
    constexpr float INV256 = 1.0f / 256.0f;
    for (int tileId = wv * 8; tileId < wv * 8 + 8; tileId++) {   // 4 m-tiles x 16 n-tiles
        int mb = (tileId >> 4) * 16;
        int nb = (tileId & 15) * 16;
        v8f acc = {};
        for (int kk = 0; kk < 32; kk++) {
            int k0 = 4 * kk + 2 * lg;
            v2f a, bb;
            a.x  = feat[(mb + lm) * 128 + k0] * INV256;      // fold pooling mean into A
            a.y  = feat[(mb + lm) * 128 + k0 + 1] * INV256;
            bb.x = fw1[(nb + lm) * 128 + k0];                // B[k,n] = fc1_w[n,k]
            bb.y = fw1[(nb + lm) * 128 + k0 + 1];
            acc = __builtin_amdgcn_wmma_f32_16x16x4_f32(
                false, a, false, bb, (short)0, acc, false, false);
        }
#pragma unroll
        for (int r = 0; r < 8; r++) {
            int bt = mb + r + 8 * lg;   // batch row
            int n  = nb + lm;           // output neuron
            float cur = acc[r] + fb1[n];
            size_t mi = (size_t)bt * 256 + n;
            float m = mem2[mi];
            float reset = (m - th2 > 0.f) ? 1.f : 0.f;
            m = BETA * m + cur - reset * th2;
            float spk = (m - th2 > 0.f) ? 1.f : 0.f;
            mem2[mi] = m;
            spk2s[bt * 256 + n] = (_Float16)spk;
            size_t rec = ((size_t)t * Bn + bt) * 256 + n;
            out[BASE_SPK2 + rec] = spk;
            out[BASE_MEM2 + rec] = m;
        }
    }
    __syncthreads();
    int idx = threadIdx.x;
    if (idx < 128) {
        int b = idx >> 1, o = idx & 1;
        float s = fb2[o];
        const float* wr = fw2 + o * 256;
        for (int k = 0; k < 256; k++) s = fmaf((float)spk2s[b * 256 + k], wr[k], s);
        float th3 = thr[3];
        float m = mem3[idx];
        float reset = (m - th3 > 0.f) ? 1.f : 0.f;
        m = BETA * m + s - reset * th3;
        float spk = (m - th3 > 0.f) ? 1.f : 0.f;
        mem3[idx] = m;
        size_t rec = ((size_t)t * Bn + b) * 2 + o;
        out[BASE_SPK3 + rec] = spk;
        out[BASE_MEM3 + rec] = m;
    }
}

extern "C" void kernel_launch(void* const* d_in, const int* in_sizes, int n_in,
                              void* d_out, int out_size, void* d_ws, size_t ws_size,
                              hipStream_t stream) {
    const float* x   = (const float*)d_in[0];
    const float* w1  = (const float*)d_in[1];
    const float* b1  = (const float*)d_in[2];
    const float* w2  = (const float*)d_in[3];
    const float* b2  = (const float*)d_in[4];
    const float* fw1 = (const float*)d_in[5];
    const float* fb1 = (const float*)d_in[6];
    const float* fw2 = (const float*)d_in[7];
    const float* fb2 = (const float*)d_in[8];
    const float* thr = (const float*)d_in[9];
    float* out = (float*)d_out;
    char* ws = (char*)d_ws;

    float*     mem0  = (float*)(ws + OFF_MEM0);
    float*     mem1  = (float*)(ws + OFF_MEM1);
    float*     mem2  = (float*)(ws + OFF_MEM2);
    float*     mem3  = (float*)(ws + OFF_MEM3);
    _Float16*  spk0h = (_Float16*)(ws + OFF_SPK0H);
    _Float16*  w2p   = (_Float16*)(ws + OFF_W2P);
    float*     feat  = (float*)(ws + OFF_FEAT);

    // Deterministic per-call init: zero all persistent state (mems + staging halos).
    size_t zero_words = ZERO_END / 4;
    zero_kernel<<<(unsigned)((zero_words + 255) / 256), 256, 0, stream>>>((float*)ws, zero_words);
    pack_w2_kernel<<<(4 * 3 * 32 * 16 + 255) / 256, 256, 0, stream>>>(w2, w2p);

    for (int t = 0; t < Tn; t++) {
        layer1_kernel<<<(Bn * 32 * L1 + 255) / 256, 256, 0, stream>>>(
            x, w1, b1, mem0, spk0h, feat, out, thr, t);
        layer2_kernel<<<Bn * 8, 256, 0, stream>>>(
            spk0h, w2p, b2, mem1, feat, out, thr, t);
        fc_fused_kernel<<<1, 256, 0, stream>>>(
            feat, fw1, fb1, fw2, fb2, mem2, mem3, out, thr, t);
    }
}
